// CausalTransformer_82566451298819
// MI455X (gfx1250) — matrix operand order repaired and checked by reading, not compiled, not run on hardware
//
#include <hip/hip_runtime.h>

// ---------------------------------------------------------------------------
// CDNA5 (gfx1250) causal transformer forward. wave32, WMMA f16->f32.
// ---------------------------------------------------------------------------

typedef __attribute__((ext_vector_type(16))) _Float16 v16h;
typedef __attribute__((ext_vector_type(8)))  _Float16 v8h;
typedef __attribute__((ext_vector_type(8)))  float    v8f;

#define WMMA(a, b, c) \
  __builtin_amdgcn_wmma_f32_16x16x32_f16(false, (a), false, (b), (short)0, (c), false, false)

// ---- fragment loaders ------------------------------------------------------
static __device__ __forceinline__ v16h load16(const _Float16* p) {
  v8h a = *(const v8h*)p;
  v8h b = *(const v8h*)(p + 8);
  v16h r;
#pragma unroll
  for (int i = 0; i < 8; ++i) { r[i] = a[i]; r[i + 8] = b[i]; }
  return r;
}
static __device__ __forceinline__ v16h combine16(v8h a, v8h b) {
  v16h r;
#pragma unroll
  for (int i = 0; i < 8; ++i) { r[i] = a[i]; r[i + 8] = b[i]; }
  return r;
}
// A-matrix 16x32 f16 fragment from row-major [16 x ld] at column k0.
static __device__ __forceinline__ v16h load_a_frag(const _Float16* A, int ld, int k0, int lane) {
  int m = lane & 15, hi = lane >> 4;
  const _Float16* p = A + (size_t)m * ld + k0 + hi * 8;
  return combine16(*(const v8h*)p, *(const v8h*)(p + 16));
}

// ---- reductions ------------------------------------------------------------
static __device__ __forceinline__ float red16_sum(float v) {
#pragma unroll
  for (int off = 1; off < 16; off <<= 1) v += __shfl_xor(v, off, 32);
  return v;
}
static __device__ __forceinline__ float red16_max(float v) {
#pragma unroll
  for (int off = 1; off < 16; off <<= 1) v = fmaxf(v, __shfl_xor(v, off, 32));
  return v;
}
static __device__ __forceinline__ float blk_sum256(float v, volatile float* sh) {
#pragma unroll
  for (int off = 16; off; off >>= 1) v += __shfl_xor(v, off, 32);
  int w = threadIdx.x >> 5;
  if ((threadIdx.x & 31) == 0) sh[w] = v;
  __syncthreads();
  float s = 0.f;
#pragma unroll
  for (int i = 0; i < 8; ++i) s += sh[i];
  __syncthreads();
  return s;
}
static __device__ __forceinline__ float blk_max256(float v, volatile float* sh) {
#pragma unroll
  for (int off = 16; off; off >>= 1) v = fmaxf(v, __shfl_xor(v, off, 32));
  int w = threadIdx.x >> 5;
  if ((threadIdx.x & 31) == 0) sh[w] = v;
  __syncthreads();
  float s = -3.0e38f;
#pragma unroll
  for (int i = 0; i < 8; ++i) s = fmaxf(s, sh[i]);
  __syncthreads();
  return s;
}

// ---------------------------------------------------------------------------
// GEMM: C[M,N] (f32) = A[M,K] (f16, row-major) @ W[K,N] (f32 weights, cvt f16)
// block = 8 waves; tile 128(M) x 64(N); K-step 32.
// W tile staged transposed in LDS: per thread two float4 global loads
// (register double-buffered across k-steps) + four packed ds_store_b32.
// ---------------------------------------------------------------------------
__global__ __launch_bounds__(256) void gemm_aw_kernel(
    const _Float16* __restrict__ A, const float* __restrict__ W,
    float* __restrict__ C, int M, int N, int K) {
  (void)M;
  __shared__ __align__(16) _Float16 lB[64][40];  // [n][k], row stride 80B
  const int tid = threadIdx.x;
  const int lane = tid & 31, wave = tid >> 5;
  const int bn = blockIdx.y * 64;
  const int row0 = blockIdx.x * 128 + wave * 16;
  const int m = lane & 15, hi = lane >> 4;

  // staging ownership: kk = 2*(tid>>4) (even), nn = 4*(tid&15)
  const int skk = (tid >> 4) * 2;
  const int snn = (tid & 15) * 4;
  const float* wp = W + (size_t)skk * N + bn + snn;
  const size_t wstep = (size_t)32 * N;

  // per-lane A base
  const _Float16* ap = A + (size_t)(row0 + m) * K + hi * 8;

  v8f acc[4];
#pragma unroll
  for (int t = 0; t < 4; ++t)
#pragma unroll
    for (int r = 0; r < 8; ++r) acc[t][r] = 0.f;

  float4 ra = *(const float4*)(wp);
  float4 rb = *(const float4*)(wp + N);

  for (int k0 = 0; k0 < K; k0 += 32) {
    // cvt + transposed store of the staged pair of W rows
    {
      union { _Float16 h[2]; unsigned u; } p0, p1, p2, p3;
      p0.h[0] = (_Float16)ra.x; p0.h[1] = (_Float16)rb.x;
      p1.h[0] = (_Float16)ra.y; p1.h[1] = (_Float16)rb.y;
      p2.h[0] = (_Float16)ra.z; p2.h[1] = (_Float16)rb.z;
      p3.h[0] = (_Float16)ra.w; p3.h[1] = (_Float16)rb.w;
      *(unsigned*)&lB[snn + 0][skk] = p0.u;
      *(unsigned*)&lB[snn + 1][skk] = p1.u;
      *(unsigned*)&lB[snn + 2][skk] = p2.u;
      *(unsigned*)&lB[snn + 3][skk] = p3.u;
    }
    __syncthreads();
    // issue next tile's global loads now; they complete under the WMMA work
    if (k0 + 32 < K) {
      wp += wstep;
      ra = *(const float4*)(wp);
      rb = *(const float4*)(wp + N);
    }
    v16h af = combine16(*(const v8h*)(ap + k0), *(const v8h*)(ap + k0 + 16));
    v16h bf[4];
#pragma unroll
    for (int t = 0; t < 4; ++t)
      bf[t] = combine16(*(const v8h*)&lB[t * 16 + m][hi * 8], *(const v8h*)&lB[t * 16 + m][16 + hi * 8]);
#pragma unroll
    for (int t = 0; t < 4; ++t)
      acc[t] = WMMA(af, bf[t], acc[t]);
    __syncthreads();
  }

#pragma unroll
  for (int t = 0; t < 4; ++t)
#pragma unroll
    for (int r = 0; r < 8; ++r) {
      int row = row0 + r + hi * 8;
      C[(size_t)row * N + bn + t * 16 + m] = acc[t][r];
    }
}

// ---------------------------------------------------------------------------
// LayerNorm (cols=1024) -> f16
// ---------------------------------------------------------------------------
__global__ __launch_bounds__(256) void ln_f16_kernel(
    const float* __restrict__ x, const float* __restrict__ g, _Float16* __restrict__ out) {
  __shared__ float sh[16];
  const size_t row = blockIdx.x;
  const float* xr = x + row * 1024;
  float v[4], s = 0.f, sq = 0.f;
#pragma unroll
  for (int i = 0; i < 4; ++i) {
    int c = threadIdx.x + i * 256;
    v[i] = xr[c]; s += v[i]; sq += v[i] * v[i];
  }
  s = blk_sum256(s, sh);
  sq = blk_sum256(sq, sh + 8);
  float mean = s * (1.f / 1024.f);
  float var = sq * (1.f / 1024.f) - mean * mean;
  float r = rsqrtf(var + 1e-5f);
#pragma unroll
  for (int i = 0; i < 4; ++i) {
    int c = threadIdx.x + i * 256;
    out[row * 1024 + c] = (_Float16)((v[i] - mean) * r * g[c]);
  }
}

// x += LayerNorm(t)*g   (attention output residual)
__global__ __launch_bounds__(256) void ln_add_kernel(
    const float* __restrict__ t, const float* __restrict__ g, float* __restrict__ x) {
  __shared__ float sh[16];
  const size_t row = blockIdx.x;
  const float* tr = t + row * 1024;
  float v[4], s = 0.f, sq = 0.f;
#pragma unroll
  for (int i = 0; i < 4; ++i) {
    int c = threadIdx.x + i * 256;
    v[i] = tr[c]; s += v[i]; sq += v[i] * v[i];
  }
  s = blk_sum256(s, sh);
  sq = blk_sum256(sq, sh + 8);
  float mean = s * (1.f / 1024.f);
  float var = sq * (1.f / 1024.f) - mean * mean;
  float r = rsqrtf(var + 1e-5f);
#pragma unroll
  for (int i = 0; i < 4; ++i) {
    int c = threadIdx.x + i * 256;
    x[row * 1024 + c] += (v[i] - mean) * r * g[c];
  }
}

// stable final LN: x/max(x) then LN -> f16
__global__ __launch_bounds__(256) void final_ln_kernel(
    const float* __restrict__ x, const float* __restrict__ g, _Float16* __restrict__ out) {
  __shared__ float sh[16];
  const size_t row = blockIdx.x;
  const float* xr = x + row * 1024;
  float v[4], mx = -3.0e38f;
#pragma unroll
  for (int i = 0; i < 4; ++i) {
    int c = threadIdx.x + i * 256;
    v[i] = xr[c]; mx = fmaxf(mx, v[i]);
  }
  mx = blk_max256(mx, sh);
  float s = 0.f, sq = 0.f;
#pragma unroll
  for (int i = 0; i < 4; ++i) { v[i] /= mx; s += v[i]; sq += v[i] * v[i]; }
  s = blk_sum256(s, sh);
  sq = blk_sum256(sq, sh + 8);
  float mean = s * (1.f / 1024.f);
  float var = sq * (1.f / 1024.f) - mean * mean;
  float r = rsqrtf(var + 1e-5f);
#pragma unroll
  for (int i = 0; i < 4; ++i) {
    int c = threadIdx.x + i * 256;
    out[row * 1024 + c] = (_Float16)((v[i] - mean) * r * g[c]);
  }
}

// ---------------------------------------------------------------------------
// q/k/v prep: rotary (first 32 dims) + l2norm, *4 (=sqrt(COS_SCALE)) on q,k.
// ---------------------------------------------------------------------------
__global__ __launch_bounds__(256) void qk_prep_kernel(
    const float* __restrict__ qpre, const float* __restrict__ kv,
    _Float16* __restrict__ qf, _Float16* __restrict__ kf, _Float16* __restrict__ vT) {
  const int row = blockIdx.x;  // b*1024 + n
  const int b = row >> 10, nn = row & 1023;
  const int lane = threadIdx.x & 31, wave = threadIdx.x >> 5;

  float cth = 1.f, sth = 0.f;
  if (lane < 16) {
    float inv = __expf(-(float)lane * 0.57564627324851142f);  // ln(10000)/16
    float th = (float)nn * inv;
    __sincosf(th, &sth, &cth);
  }

#pragma unroll
  for (int hh = 0; hh < 2; ++hh) {
    int h = wave * 2 + hh;
    const float* src = qpre + (size_t)row * 1024 + h * 64;
    float x0 = src[2 * lane], x1 = src[2 * lane + 1];
    float y0 = x0, y1 = x1;
    if (lane < 16) { y0 = x0 * cth - x1 * sth; y1 = x1 * cth + x0 * sth; }
    float ss = y0 * y0 + y1 * y1;
#pragma unroll
    for (int off = 16; off; off >>= 1) ss += __shfl_xor(ss, off, 32);
    float sc = 4.f / fmaxf(sqrtf(ss), 1e-12f);
    _Float16* dst = qf + ((((size_t)b * 16 + h) * 1024) + nn) * 64;
    dst[2 * lane] = (_Float16)(y0 * sc);
    dst[2 * lane + 1] = (_Float16)(y1 * sc);
  }

  if (wave == 0) {  // k
    const float* src = kv + (size_t)row * 128;
    float x0 = src[2 * lane], x1 = src[2 * lane + 1];
    float y0 = x0, y1 = x1;
    if (lane < 16) { y0 = x0 * cth - x1 * sth; y1 = x1 * cth + x0 * sth; }
    float ss = y0 * y0 + y1 * y1;
#pragma unroll
    for (int off = 16; off; off >>= 1) ss += __shfl_xor(ss, off, 32);
    float sc = 4.f / fmaxf(sqrtf(ss), 1e-12f);
    _Float16* dst = kf + ((size_t)b * 1056 + nn + 1) * 64;
    dst[2 * lane] = (_Float16)(y0 * sc);
    dst[2 * lane + 1] = (_Float16)(y1 * sc);
  } else if (wave == 1) {  // v -> transposed [d][j]
    const float* src = kv + (size_t)row * 128 + 64;
    vT[((size_t)b * 64 + 2 * lane) * 1056 + nn + 1] = (_Float16)src[2 * lane];
    vT[((size_t)b * 64 + 2 * lane + 1) * 1056 + nn + 1] = (_Float16)src[2 * lane + 1];
  }
}

// null kv row (j=0) + zero padding rows/cols 1025..1055. grid=B, 64 threads.
__global__ void null_pad_kernel(const float* __restrict__ nkv,
                                _Float16* __restrict__ kf, _Float16* __restrict__ vT) {
  __shared__ float sh[2];
  const int b = blockIdx.x, t = threadIdx.x;
  float v = nkv[t];
  float ss = v * v;
#pragma unroll
  for (int off = 16; off; off >>= 1) ss += __shfl_xor(ss, off, 32);
  if ((t & 31) == 0) sh[t >> 5] = ss;
  __syncthreads();
  float sc = 4.f / fmaxf(sqrtf(sh[0] + sh[1]), 1e-12f);
  kf[(size_t)b * 1056 * 64 + t] = (_Float16)(v * sc);
  vT[((size_t)b * 64 + t) * 1056] = (_Float16)nkv[64 + t];
  for (int idx = t; idx < 31 * 64; idx += 64)
    kf[((size_t)b * 1056 + 1025) * 64 + idx] = (_Float16)0.f;
  for (int i = 0; i < 31; ++i)
    vT[((size_t)b * 64 + t) * 1056 + 1025 + i] = (_Float16)0.f;
}

// relative-position bias table tab[h][n], n = clamp(i-j,0,1023)
__global__ void bias_tab_kernel(const float* __restrict__ emb, float* __restrict__ tab) {
  int idx = blockIdx.x * 256 + threadIdx.x;  // 16*1024
  int h = idx >> 10, n = idx & 1023;
  int bucket;
  if (n < 16) bucket = n;
  else {
    float nf = (float)n;
    int vl = 16 + (int)(__logf(nf * (1.f / 16.f)) * (16.f / __logf(8.f)));
    bucket = vl < 31 ? vl : 31;
  }
  tab[idx] = emb[bucket * 16 + h];
}

// ---------------------------------------------------------------------------
// Flash-style attention. One wave = 16 query rows of one (b,h).
// block = 4 waves; grid = B*H*(N/16)/4 = 512.
// ---------------------------------------------------------------------------
__global__ __launch_bounds__(128) void attn_kernel(
    const _Float16* __restrict__ qf, const _Float16* __restrict__ kf,
    const _Float16* __restrict__ vT, const float* __restrict__ btab,
    _Float16* __restrict__ O) {
  __shared__ __align__(16) _Float16 lP[4][16][40];
  const int tid = threadIdx.x;
  const int lane = tid & 31, wave = tid >> 5;
  const int gw = blockIdx.x * 4 + wave;
  const int i0 = (gw & 63) * 16;
  const int h = (gw >> 6) & 15;
  const int b = gw >> 10;
  const int m = lane & 15, hi = lane >> 4;

  const _Float16* qbase = qf + (((size_t)b * 16 + h) * 1024 + i0) * 64;
  const v16h qa0 = load_a_frag(qbase, 64, 0, lane);
  const v16h qa1 = load_a_frag(qbase, 64, 32, lane);
  const _Float16* kbase = kf + (size_t)b * 1056 * 64 + (size_t)m * 64 + hi * 8;
  const _Float16* vbase = vT + (size_t)b * 64 * 1056 + (size_t)m * 1056 + 16 * hi;
  const float* bt = btab + h * 1024;

  float rmax[8], rsum[8];
  v8f o[4];
#pragma unroll
  for (int r = 0; r < 8; ++r) { rmax[r] = -3.0e38f; rsum[r] = 0.f; }
#pragma unroll
  for (int t = 0; t < 4; ++t)
#pragma unroll
    for (int r = 0; r < 8; ++r) o[t][r] = 0.f;

  int jmax = i0 + 17; if (jmax > 1025) jmax = 1025;
  for (int j0 = 0; j0 < jmax; j0 += 32) {
    // ---- S = Q K^T : load all K fragments, then issue WMMAs ----
    v16h kb0[2], kb1[2];
#pragma unroll
    for (int ns = 0; ns < 2; ++ns) {
      const _Float16* kp = kbase + (size_t)(j0 + ns * 16) * 64;
      kb0[ns] = combine16(*(const v8h*)kp, *(const v8h*)(kp + 16));
      kb1[ns] = combine16(*(const v8h*)(kp + 32), *(const v8h*)(kp + 48));
    }
    v8f s[2];
#pragma unroll
    for (int ns = 0; ns < 2; ++ns) {
      v8f c;
#pragma unroll
      for (int r = 0; r < 8; ++r) c[r] = 0.f;
      c = WMMA(qa0, kb0[ns], c);
      c = WMMA(qa1, kb1[ns], c);
      s[ns] = c;
    }
    // ---- V fragments issued early (overlap with softmax + LDS) ----
    v16h vfr[4];
#pragma unroll
    for (int t = 0; t < 4; ++t)
      vfr[t] = load16(vbase + (size_t)(16 * t) * 1056 + j0);
    // ---- bias + causal mask + online softmax ----
#pragma unroll
    for (int r = 0; r < 8; ++r) {
      const int i = i0 + r + 8 * hi;
#pragma unroll
      for (int ns = 0; ns < 2; ++ns) {
        int j = j0 + ns * 16 + m;
        float val = s[ns][r];
        int d = i - j; d = d < 0 ? 0 : (d > 1023 ? 1023 : d);
        val += bt[d];
        if (j >= 1025 || j > i + 1) val = -1.0e30f;
        s[ns][r] = val;
      }
      float lm = red16_max(fmaxf(s[0][r], s[1][r]));
      float nm = fmaxf(rmax[r], lm);
      float corr = __expf(rmax[r] - nm);
      rmax[r] = nm;
      float p0 = __expf(s[0][r] - nm);
      float p1 = __expf(s[1][r] - nm);
      rsum[r] = rsum[r] * corr + red16_sum(p0 + p1);
      lP[wave][r + 8 * hi][m]      = (_Float16)p0;
      lP[wave][r + 8 * hi][16 + m] = (_Float16)p1;
#pragma unroll
      for (int t = 0; t < 4; ++t) o[t][r] *= corr;
    }
    // ---- O += P V  (P 16x32 A-frag re-read from LDS; in-order DS) ----
    const _Float16* pp = &lP[wave][m][0];
    v16h pfrag = combine16(*(const v8h*)(pp + 16 * hi),
                           *(const v8h*)(pp + 16 * hi + 8));
#pragma unroll
    for (int t = 0; t < 4; ++t)
      o[t] = WMMA(pfrag, vfr[t], o[t]);
  }

  // ---- finalize: divide by row sums, store to [b, n, h*64+d] f16 ----
#pragma unroll
  for (int t = 0; t < 4; ++t)
#pragma unroll
    for (int r = 0; r < 8; ++r) {
      int i = i0 + r + 8 * hi;
      int d = 16 * t + m;
      O[((size_t)b * 1024 + i) * 1024 + h * 64 + d] = (_Float16)(o[t][r] / rsum[r]);
    }
}

// a * silu(g) -> f16 ; h [rows,8192] -> out [rows,4096]
__global__ void gate_kernel(const float* __restrict__ h, _Float16* __restrict__ out, size_t n) {
  size_t idx = (size_t)blockIdx.x * 256 + threadIdx.x;
  if (idx < n) {
    size_t r = idx >> 12; int c = (int)(idx & 4095);
    float a = h[r * 8192 + c];
    float g = h[r * 8192 + 4096 + c];
    out[idx] = (_Float16)(a * g / (1.f + __expf(-g)));
  }
}

__global__ void add_inplace_kernel(float* __restrict__ x, const float* __restrict__ y) {
  size_t idx = (size_t)blockIdx.x * 256 + threadIdx.x;
  x[idx] += y[idx];
}

// ---------------------------------------------------------------------------
extern "C" void kernel_launch(void* const* d_in, const int* in_sizes, int n_in,
                              void* d_out, int out_size, void* d_ws, size_t ws_size,
                              hipStream_t stream) {
  (void)in_sizes; (void)n_in; (void)out_size; (void)ws_size;
  const int ROWS = 2048;  // B*N
  char* ws = (char*)d_ws;
  float*    xbuf = (float*)   (ws + 0);
  _Float16* hf   = (_Float16*)(ws + 8388608u);
  float*    qpre = (float*)   (ws + 12582912u);
  float*    kvb  = (float*)   (ws + 20971520u);
  _Float16* qf   = (_Float16*)(ws + 22020096u);
  _Float16* kf   = (_Float16*)(ws + 26214400u);
  _Float16* vT   = (_Float16*)(ws + 26484736u);
  _Float16* aO   = (_Float16*)(ws + 26755072u);
  float*    tbuf = (float*)   (ws + 30949376u);
  float*    ffh  = (float*)   (ws + 39337984u);
  _Float16* gbuf = (_Float16*)(ws + 106446848u);
  float*    btab = (float*)   (ws + 123224064u);

  const float* x_in  = (const float*)d_in[0];
  const float* emb   = (const float*)d_in[37];
  const float* fin_g = (const float*)d_in[38];
  const float* wproj = (const float*)d_in[39];

  hipMemcpyAsync(xbuf, x_in, (size_t)ROWS * 1024 * sizeof(float),
                 hipMemcpyDeviceToDevice, stream);
  bias_tab_kernel<<<64, 256, 0, stream>>>(emb, btab);

  for (int l = 0; l < 4; ++l) {
    const int base = 1 + l * 9;
    const float* attn_g = (const float*)d_in[base + 0];
    const float* wq     = (const float*)d_in[base + 1];
    const float* wkv    = (const float*)d_in[base + 2];
    const float* nkv    = (const float*)d_in[base + 3];
    const float* wo     = (const float*)d_in[base + 4];
    const float* out_g  = (const float*)d_in[base + 5];
    const float* ff_g   = (const float*)d_in[base + 6];
    const float* w1     = (const float*)d_in[base + 7];
    const float* w2     = (const float*)d_in[base + 8];

    // ---- attention ----
    ln_f16_kernel<<<ROWS, 256, 0, stream>>>(xbuf, attn_g, hf);
    gemm_aw_kernel<<<dim3(16, 16), 256, 0, stream>>>(hf, wq, qpre, ROWS, 1024, 1024);
    gemm_aw_kernel<<<dim3(16, 2), 256, 0, stream>>>(hf, wkv, kvb, ROWS, 128, 1024);
    qk_prep_kernel<<<ROWS, 256, 0, stream>>>(qpre, kvb, qf, kf, vT);
    null_pad_kernel<<<2, 64, 0, stream>>>(nkv, kf, vT);
    attn_kernel<<<512, 128, 0, stream>>>(qf, kf, vT, btab, aO);
    gemm_aw_kernel<<<dim3(16, 16), 256, 0, stream>>>(aO, wo, tbuf, ROWS, 1024, 1024);
    ln_add_kernel<<<ROWS, 256, 0, stream>>>(tbuf, out_g, xbuf);

    // ---- feed-forward ----
    ln_f16_kernel<<<ROWS, 256, 0, stream>>>(xbuf, ff_g, hf);
    gemm_aw_kernel<<<dim3(16, 128), 256, 0, stream>>>(hf, w1, ffh, ROWS, 8192, 1024);
    gate_kernel<<<32768, 256, 0, stream>>>(ffh, gbuf, (size_t)ROWS * 4096);
    gemm_aw_kernel<<<dim3(16, 16), 256, 0, stream>>>(gbuf, w2, tbuf, ROWS, 1024, 4096);
    add_inplace_kernel<<<8192, 256, 0, stream>>>(xbuf, tbuf);
  }

  final_ln_kernel<<<ROWS, 256, 0, stream>>>(xbuf, fin_g, hf);
  gemm_aw_kernel<<<dim3(16, 16), 256, 0, stream>>>(hf, wproj, (float*)d_out,
                                                   ROWS, 1024, 1024);
}
